// multihead_self_attention_55181739819726
// MI455X (gfx1250) — matrix operand order repaired and checked by reading, not compiled, not run on hardware
//
#include <hip/hip_runtime.h>
#include <hip/hip_bf16.h>
#include <math.h>

typedef __bf16 bf16_t;
typedef unsigned int u32;
typedef unsigned long long u64;
typedef __attribute__((ext_vector_type(16))) __bf16 v16bf;
typedef __attribute__((ext_vector_type(8)))  __bf16 v8bf;
typedef __attribute__((ext_vector_type(8)))  float  v8f;
typedef __attribute__((ext_vector_type(4)))  u32    v4u;
typedef __attribute__((ext_vector_type(8)))  u32    v8u;

#define DM    1024
#define HEADS 16
#define DK    64
#define BATCH 4
#define SEQ   2048
#define TOK   (BATCH * SEQ)

// ---------------------------------------------------------------------------
// WMMA helper (CDNA5 wave32, V_WMMA_F32_16X16X32_BF16)
// ---------------------------------------------------------------------------
static __device__ __forceinline__ v8f wmma_bf16(v16bf a, v16bf b, v8f c) {
  return __builtin_amdgcn_wmma_f32_16x16x32_bf16(false, a, false, b, (short)0, c,
                                                 false, false);
}

// A fragment: 16(M) x 32(K) bf16 from row-major source (ld elems).
// lanes 0-15: row=lane, K={0..7,16..23}; lanes 16-31: row=lane-16, K={8..15,24..31}
static __device__ __forceinline__ v16bf load_a_frag(const bf16_t* src, int ld) {
  const int lane = threadIdx.x & 31;
  const bf16_t* p = src + (lane & 15) * ld + ((lane < 16) ? 0 : 8);
  v8bf lo = *(const v8bf*)(p);
  v8bf hi = *(const v8bf*)(p + 16);
  v16bf f;
#pragma unroll
  for (int i = 0; i < 8; ++i) { f[i] = lo[i]; f[i + 8] = hi[i]; }
  return f;
}

// B fragment: 32(K) x 16(N) bf16 from N-major source (row = n, K contiguous).
// lanes 0-15: col=lane, K={0..7,16..23}; lanes 16-31: col=lane-16, K={8..15,24..31}
static __device__ __forceinline__ v16bf load_b_frag(const bf16_t* src, int ld) {
  const int lane = threadIdx.x & 31;
  const bf16_t* p = src + (lane & 15) * ld + ((lane < 16) ? 0 : 8);
  v8bf lo = *(const v8bf*)(p);
  v8bf hi = *(const v8bf*)(p + 16);
  v16bf f;
#pragma unroll
  for (int i = 0; i < 8; ++i) { f[i] = lo[i]; f[i + 8] = hi[i]; }
  return f;
}

// ---------------------------------------------------------------------------
// Tensor Data Mover: async 2-D tile load Global -> LDS (bf16 elements).
// D# per CDNA5 ISA. pad_interval/pad_amount make the
// TDM insert the LDS row padding the fragment loaders expect.
// ---------------------------------------------------------------------------
static __device__ __forceinline__ void tdm_load_2d(u32 lds_byte_addr,
                                                   const void* gptr,
                                                   u32 tensor_d0, u32 tensor_d1,
                                                   u64 stride0_elems,
                                                   u32 tile_d0, u32 tile_d1,
                                                   u32 pad_interval,
                                                   u32 pad_amount) {
  u64 ga = (u64)(uintptr_t)gptr;
  v4u g0;
  g0.x = 1u;                                  // count=1 (valid), no gather
  g0.y = lds_byte_addr;                       // LDS byte address
  g0.z = (u32)ga;                             // global_addr[31:0]
  g0.w = (u32)(ga >> 32) | (2u << 30);        // global_addr[56:32] | type=2
  v8u g1;
  g1.s0 = (1u << 16)                          // data_size = 1 -> 2 bytes
        | (1u << 20)                          // pad_enable
        | (pad_interval << 22) | (pad_amount << 25);
  g1.s1 = (tensor_d0 & 0xffffu) << 16;        // tensor_dim0[15:0] (abar=0)
  g1.s2 = (tensor_d0 >> 16) | ((tensor_d1 & 0xffffu) << 16);
  g1.s3 = (tensor_d1 >> 16) | (tile_d0 << 16);
  g1.s4 = tile_d1;                            // tile_dim2 = 0 (2-D)
  g1.s5 = (u32)stride0_elems;                 // tensor_dim0_stride[31:0]
  g1.s6 = (u32)(stride0_elems >> 32);         // stride0[47:32], stride1=0
  g1.s7 = 0u;
  asm volatile("tensor_load_to_lds %0, %1" :: "s"(g0), "s"(g1) : "memory");
}

// ---------------------------------------------------------------------------
// fp32 -> bf16 conversion
// ---------------------------------------------------------------------------
__global__ void cvt_f32_bf16(const float* __restrict__ src,
                             bf16_t* __restrict__ dst, int n) {
  int i = blockIdx.x * blockDim.x + threadIdx.x;
  if (i < n) dst[i] = (bf16_t)src[i];
}

// ---------------------------------------------------------------------------
// C = A * B^T, A: MxK row-major bf16, B: NxK row-major bf16.
// WG tile 256(M) x 64(N), K panel 32, double-buffered LDS, 8 waves,
// wave w owns rows w*32..w*32+31 (2 A-frags, 8 C-frags).
// Output element type is a template parameter -> branch-free epilogue.
// ---------------------------------------------------------------------------
#define GM 256
#define GN 64
#define GK 32
#define GPAD 8
#define GLDA (GK + GPAD)
template <typename OutT>
__global__ __launch_bounds__(256) void gemm_nt_bf16(
    const bf16_t* __restrict__ A, const bf16_t* __restrict__ B,
    OutT* __restrict__ C, int M, int N, int K) {
  __shared__ bf16_t At[2][GM][GLDA];
  __shared__ bf16_t Bt[2][GN][GLDA];
  const int t = threadIdx.x, lane = t & 31, w = t >> 5;
  const int row0 = blockIdx.y * GM, col0 = blockIdx.x * GN;
  const bf16_t* Arow = A + (size_t)(row0 + t) * K;   // thread t copies A row t
  const bf16_t* Brow = B + (size_t)(col0 + (t & (GN - 1))) * K;

  // preload K-panel 0
  {
    *(v8bf*)&At[0][t][0]  = *(const v8bf*)(Arow + 0);
    *(v8bf*)&At[0][t][8]  = *(const v8bf*)(Arow + 8);
    *(v8bf*)&At[0][t][16] = *(const v8bf*)(Arow + 16);
    *(v8bf*)&At[0][t][24] = *(const v8bf*)(Arow + 24);
    if (t < GN) {
      *(v8bf*)&Bt[0][t][0]  = *(const v8bf*)(Brow + 0);
      *(v8bf*)&Bt[0][t][8]  = *(const v8bf*)(Brow + 8);
      *(v8bf*)&Bt[0][t][16] = *(const v8bf*)(Brow + 16);
      *(v8bf*)&Bt[0][t][24] = *(const v8bf*)(Brow + 24);
    }
  }
  __syncthreads();

  const int nk = K / GK;
  v8f acc[2][4] = {};
  for (int i = 0; i < nk; ++i) {
    const int cur = i & 1;
    const bool pf = (i + 1 < nk);
    v8bf a0, a1, a2, a3, b0, b1, b2, b3;
    if (pf) {  // prefetch next panel into registers (overlaps with WMMA)
      const bf16_t* s = Arow + (i + 1) * GK;
      a0 = *(const v8bf*)(s + 0);  a1 = *(const v8bf*)(s + 8);
      a2 = *(const v8bf*)(s + 16); a3 = *(const v8bf*)(s + 24);
      if (t < GN) {
        const bf16_t* sb = Brow + (i + 1) * GK;
        b0 = *(const v8bf*)(sb + 0);  b1 = *(const v8bf*)(sb + 8);
        b2 = *(const v8bf*)(sb + 16); b3 = *(const v8bf*)(sb + 24);
      }
    }
    // all fragments live before the WMMA chain -> overlapped ds_loads
    v16bf fa0 = load_a_frag(&At[cur][w * 32][0],      GLDA);
    v16bf fa1 = load_a_frag(&At[cur][w * 32 + 16][0], GLDA);
    v16bf fb0 = load_b_frag(&Bt[cur][0][0],  GLDA);
    v16bf fb1 = load_b_frag(&Bt[cur][16][0], GLDA);
    v16bf fb2 = load_b_frag(&Bt[cur][32][0], GLDA);
    v16bf fb3 = load_b_frag(&Bt[cur][48][0], GLDA);
    acc[0][0] = wmma_bf16(fa0, fb0, acc[0][0]);
    acc[0][1] = wmma_bf16(fa0, fb1, acc[0][1]);
    acc[0][2] = wmma_bf16(fa0, fb2, acc[0][2]);
    acc[0][3] = wmma_bf16(fa0, fb3, acc[0][3]);
    acc[1][0] = wmma_bf16(fa1, fb0, acc[1][0]);
    acc[1][1] = wmma_bf16(fa1, fb1, acc[1][1]);
    acc[1][2] = wmma_bf16(fa1, fb2, acc[1][2]);
    acc[1][3] = wmma_bf16(fa1, fb3, acc[1][3]);
    if (pf) {  // stores go to the other buffer: no extra barrier needed
      *(v8bf*)&At[cur ^ 1][t][0]  = a0;
      *(v8bf*)&At[cur ^ 1][t][8]  = a1;
      *(v8bf*)&At[cur ^ 1][t][16] = a2;
      *(v8bf*)&At[cur ^ 1][t][24] = a3;
      if (t < GN) {
        *(v8bf*)&Bt[cur ^ 1][t][0]  = b0;
        *(v8bf*)&Bt[cur ^ 1][t][8]  = b1;
        *(v8bf*)&Bt[cur ^ 1][t][16] = b2;
        *(v8bf*)&Bt[cur ^ 1][t][24] = b3;
      }
    }
    __syncthreads();
  }
  // C layout: VGPR e holds (M = e + 8*(lane>=16), N = lane&15)
  const int cbase = col0 + (lane & 15);
#pragma unroll
  for (int mi = 0; mi < 2; ++mi) {
    const int rbase = row0 + w * 32 + mi * 16 + ((lane < 16) ? 0 : 8);
#pragma unroll
    for (int n = 0; n < 4; ++n)
#pragma unroll
      for (int e = 0; e < 8; ++e) {
        size_t idx = (size_t)(rbase + e) * N + cbase + n * 16;
        C[idx] = (OutT)acc[mi][n][e];
      }
  }
}

// ---------------------------------------------------------------------------
// RoPE on Q,K (fp32 math) with permute [b,s,h,d] -> [b,h,s,d]
// ---------------------------------------------------------------------------
__global__ void rope_qk(const bf16_t* __restrict__ Qlin,
                        const bf16_t* __restrict__ Klin,
                        const int* __restrict__ pos,
                        bf16_t* __restrict__ Qr, bf16_t* __restrict__ Kr) {
  int idx = blockIdx.x * blockDim.x + threadIdx.x;
  if (idx >= TOK * (DM / 2)) return;
  int bs = idx / (DM / 2);
  int p  = idx % (DM / 2);
  int h = p / (DK / 2), i = p % (DK / 2);
  int b = bs / SEQ, s = bs % SEQ;
  float ang = (float)pos[s] *
              __expf(-(2.0f * (float)i / (float)DK) * 9.210340371976184f);
  float sn, cs;
  __sincosf(ang, &sn, &cs);
  size_t src = (size_t)bs * DM + h * DK + 2 * i;
  size_t dst = (((size_t)b * HEADS + h) * SEQ + s) * DK + 2 * i;
  {
    float e = (float)Qlin[src], o = (float)Qlin[src + 1];
    Qr[dst]     = (bf16_t)(cs * e - sn * o);
    Qr[dst + 1] = (bf16_t)(sn * e + cs * o);
  }
  {
    float e = (float)Klin[src], o = (float)Klin[src + 1];
    Kr[dst]     = (bf16_t)(cs * e - sn * o);
    Kr[dst + 1] = (bf16_t)(sn * e + cs * o);
  }
}

// V: [b,s,h,d] -> Vt [b,h,d,s] so PV GEMM's B operand is seq-contiguous.
__global__ void scatter_v(const bf16_t* __restrict__ Vlin,
                          bf16_t* __restrict__ Vt) {
  int idx = blockIdx.x * blockDim.x + threadIdx.x;
  if (idx >= TOK * DM) return;
  int bs = idx / DM, p = idx % DM;
  int h = p / DK, d = p % DK;
  int b = bs / SEQ, s = bs % SEQ;
  Vt[(((size_t)b * HEADS + h) * DK + d) * SEQ + s] = Vlin[idx];
}

// ---------------------------------------------------------------------------
// Flash attention: WG = 128 query rows (8 waves x 16), key tiles of 32.
// K/V tiles fed by the Tensor Data Mover (double-buffered, wave 0 issues,
// s_wait_tensorcnt + one workgroup barrier per tile publishes the data).
// Online softmax kept entirely in WMMA C-layout via 16-lane-half shfl_xor.
// ---------------------------------------------------------------------------
#define QTB 128
#define KTB 32
#define KP  8
__global__ __launch_bounds__(256) void attn_kernel(
    const bf16_t* __restrict__ Q, const bf16_t* __restrict__ K,
    const bf16_t* __restrict__ Vt, bf16_t* __restrict__ AO) {
  __shared__ bf16_t kbuf[2][KTB][DK + KP];     // 32 key rows x 64 d (+pad)
  __shared__ bf16_t vbuf[2][DK][KTB + KP];     // 64 d rows x 32 seq (+pad)
  __shared__ bf16_t pbuf[8][16][KTB + KP];     // per-wave P tile (16x32)
  const int t = threadIdx.x, lane = t & 31, w = t >> 5;
  const int bh = blockIdx.y;                   // b*16 + h
  const int qbase = blockIdx.x * QTB;
  const bf16_t* Qh = Q  + (size_t)bh * SEQ * DK;
  const bf16_t* Kh = K  + (size_t)bh * SEQ * DK;
  const bf16_t* Vh = Vt + (size_t)bh * DK * SEQ;

  // TDM issue for key tile starting at kt into buffer `buf`.
  // K rows are 64 bf16 = 32 dwords -> pad_interval=4 (32 dw), pad_amount=3 (4 dw).
  // V rows are 32 bf16 = 16 dwords -> pad_interval=3 (16 dw), pad_amount=3.
  auto issue_tiles = [&](int kt, int buf) {
    tdm_load_2d((u32)(uintptr_t)(const void*)&kbuf[buf][0][0],
                Kh + (size_t)kt * DK, DK, SEQ, DK, DK, KTB, 4u, 3u);
    tdm_load_2d((u32)(uintptr_t)(const void*)&vbuf[buf][0][0],
                Vh + kt, SEQ, DK, SEQ, KTB, DK, 3u, 3u);
  };

  // Q fragments (16 rows x 64 d), kept in registers for the whole pass
  const bf16_t* qrow = Qh + (size_t)(qbase + w * 16) * DK;
  v16bf qa0 = load_a_frag(qrow, DK);
  v16bf qa1 = load_a_frag(qrow + 32, DK);

  v8f acc[4] = {};
  float m[8], l[8];
#pragma unroll
  for (int e = 0; e < 8; ++e) { m[e] = -3.0e38f; l[e] = 0.0f; }

  const int myrow0 = qbase + w * 16 + ((lane < 16) ? 0 : 8);
  const int nt = (qbase + QTB) / KTB;          // causal: keys [0, qbase+128)

  if (w == 0) issue_tiles(0, 0);

  for (int i = 0; i < nt; ++i) {
    const int kt = i * KTB, buf = i & 1;
    if (w == 0) __builtin_amdgcn_s_wait_tensorcnt(0);
    __syncthreads();                            // publish tile i
    if (w == 0 && i + 1 < nt) issue_tiles(kt + KTB, buf ^ 1);  // overlap DMA

    // ---- S = Q K^T for two 16-wide key subtiles (chained K=32 WMMAs) ----
    v16bf b00 = load_b_frag(&kbuf[buf][0][0],   DK + KP);
    v16bf b01 = load_b_frag(&kbuf[buf][0][32],  DK + KP);
    v16bf b10 = load_b_frag(&kbuf[buf][16][0],  DK + KP);
    v16bf b11 = load_b_frag(&kbuf[buf][16][32], DK + KP);
    v8f s0 = {}, s1 = {};
    s0 = wmma_bf16(qa0, b00, s0);
    s0 = wmma_bf16(qa1, b01, s0);
    s1 = wmma_bf16(qa0, b10, s1);
    s1 = wmma_bf16(qa1, b11, s1);

    // ---- scale + causal mask ----
    const int col = kt + (lane & 15);
#pragma unroll
    for (int e = 0; e < 8; ++e) {
      int row = myrow0 + e;
      float a0 = s0[e] * 0.125f;  // 1/sqrt(64)
      float a1 = s1[e] * 0.125f;
      if (col > row)      a0 = -3.0e38f;
      if (col + 16 > row) a1 = -3.0e38f;
      s0[e] = a0; s1[e] = a1;
    }

    // ---- row stats across the 16-lane half that owns this row ----
    float tmax[8], sc[8], rs[8];
#pragma unroll
    for (int e = 0; e < 8; ++e) tmax[e] = fmaxf(s0[e], s1[e]);
#pragma unroll
    for (int x = 1; x <= 8; x <<= 1)
#pragma unroll
      for (int e = 0; e < 8; ++e)
        tmax[e] = fmaxf(tmax[e], __shfl_xor(tmax[e], x, 32));
#pragma unroll
    for (int e = 0; e < 8; ++e) {
      float mn = fmaxf(m[e], tmax[e]);
      sc[e] = __expf(m[e] - mn);
      m[e]  = mn;
      s0[e] = __expf(s0[e] - mn);
      s1[e] = __expf(s1[e] - mn);
      rs[e] = s0[e] + s1[e];
    }
#pragma unroll
    for (int x = 1; x <= 8; x <<= 1)
#pragma unroll
      for (int e = 0; e < 8; ++e) rs[e] += __shfl_xor(rs[e], x, 32);
#pragma unroll
    for (int e = 0; e < 8; ++e) l[e] = l[e] * sc[e] + rs[e];

    // ---- rescale O accumulators (same C-layout -> per-element row scale) ----
#pragma unroll
    for (int n = 0; n < 4; ++n)
#pragma unroll
      for (int e = 0; e < 8; ++e) acc[n][e] *= sc[e];

    // ---- P (C-layout) -> per-wave LDS -> A fragment ----
    {
      int prow = (lane < 16) ? 0 : 8;
      int pcol = lane & 15;
#pragma unroll
      for (int e = 0; e < 8; ++e) {
        pbuf[w][prow + e][pcol]      = (bf16_t)s0[e];
        pbuf[w][prow + e][pcol + 16] = (bf16_t)s1[e];
      }
    }
    v16bf pa  = load_a_frag(&pbuf[w][0][0], KTB + KP);
    v16bf vb0 = load_b_frag(&vbuf[buf][0][0],  KTB + KP);
    v16bf vb1 = load_b_frag(&vbuf[buf][16][0], KTB + KP);
    v16bf vb2 = load_b_frag(&vbuf[buf][32][0], KTB + KP);
    v16bf vb3 = load_b_frag(&vbuf[buf][48][0], KTB + KP);
    acc[0] = wmma_bf16(pa, vb0, acc[0]);
    acc[1] = wmma_bf16(pa, vb1, acc[1]);
    acc[2] = wmma_bf16(pa, vb2, acc[2]);
    acc[3] = wmma_bf16(pa, vb3, acc[3]);
  }

  // ---- epilogue: normalize, write AO[b, s, h*64 + d] ----
  const int b = bh >> 4, h = bh & 15;
#pragma unroll
  for (int n = 0; n < 4; ++n)
#pragma unroll
    for (int e = 0; e < 8; ++e) {
      int row = myrow0 + e;
      size_t o = ((size_t)b * SEQ + row) * DM + h * DK + n * 16 + (lane & 15);
      AO[o] = (bf16_t)(acc[n][e] / l[e]);
    }
}

// ---------------------------------------------------------------------------
// Host-side orchestration
// ---------------------------------------------------------------------------
extern "C" void kernel_launch(void* const* d_in, const int* in_sizes, int n_in,
                              void* d_out, int out_size, void* d_ws, size_t ws_size,
                              hipStream_t stream) {
  const float* x   = (const float*)d_in[0];
  const float* Wq  = (const float*)d_in[1];
  const float* Wk  = (const float*)d_in[2];
  const float* Wv  = (const float*)d_in[3];
  const float* Wo  = (const float*)d_in[4];
  const int*   pos = (const int*)d_in[5];
  float* out = (float*)d_out;

  char* ws = (char*)d_ws;
  size_t off = 0;
  bf16_t* Xbf  = (bf16_t*)(ws + off); off += (size_t)TOK * DM * sizeof(bf16_t);
  bf16_t* Wqb  = (bf16_t*)(ws + off); off += (size_t)DM * DM * sizeof(bf16_t);
  bf16_t* Wkb  = (bf16_t*)(ws + off); off += (size_t)DM * DM * sizeof(bf16_t);
  bf16_t* Wvb  = (bf16_t*)(ws + off); off += (size_t)DM * DM * sizeof(bf16_t);
  bf16_t* Wob  = (bf16_t*)(ws + off); off += (size_t)DM * DM * sizeof(bf16_t);
  bf16_t* Qlin = (bf16_t*)(ws + off); off += (size_t)TOK * DM * sizeof(bf16_t);
  bf16_t* Klin = (bf16_t*)(ws + off); off += (size_t)TOK * DM * sizeof(bf16_t);
  bf16_t* Vlin = (bf16_t*)(ws + off); off += (size_t)TOK * DM * sizeof(bf16_t);
  // Aliases (all uses strictly ordered on `stream`):
  bf16_t* Qr = (bf16_t*)out;                // d_out as scratch; dead before final GEMM
  bf16_t* Kr = (bf16_t*)out + (size_t)TOK * DM;
  bf16_t* Vt = Xbf;                         // Xbf dead after projections
  bf16_t* AO = Qlin;                        // Qlin dead after RoPE

  const int nX = TOK * DM, nW = DM * DM;
  cvt_f32_bf16<<<(nX + 255) / 256, 256, 0, stream>>>(x,  Xbf, nX);
  cvt_f32_bf16<<<(nW + 255) / 256, 256, 0, stream>>>(Wq, Wqb, nW);
  cvt_f32_bf16<<<(nW + 255) / 256, 256, 0, stream>>>(Wk, Wkb, nW);
  cvt_f32_bf16<<<(nW + 255) / 256, 256, 0, stream>>>(Wv, Wvb, nW);
  cvt_f32_bf16<<<(nW + 255) / 256, 256, 0, stream>>>(Wo, Wob, nW);

  dim3 gp(DM / GN, TOK / GM);  // (16, 32)
  gemm_nt_bf16<bf16_t><<<gp, 256, 0, stream>>>(Xbf, Wqb, Qlin, TOK, DM, DM);
  gemm_nt_bf16<bf16_t><<<gp, 256, 0, stream>>>(Xbf, Wkb, Klin, TOK, DM, DM);
  gemm_nt_bf16<bf16_t><<<gp, 256, 0, stream>>>(Xbf, Wvb, Vlin, TOK, DM, DM);

  const int nP = TOK * (DM / 2);
  rope_qk<<<(nP + 255) / 256, 256, 0, stream>>>(Qlin, Klin, pos, Qr, Kr);
  scatter_v<<<(nX + 255) / 256, 256, 0, stream>>>(Vlin, Vt);

  dim3 ga(SEQ / QTB, BATCH * HEADS);  // (16, 64)
  attn_kernel<<<ga, 256, 0, stream>>>(Qr, Kr, Vt, AO);

  gemm_nt_bf16<float><<<gp, 256, 0, stream>>>(AO, Wob, out, TOK, DM, DM);
}
